// DynamicIterativeRefiner_16406775070826
// MI455X (gfx1250) — matrix operand order repaired and checked by reading, not compiled, not run on hardware
//
#include <hip/hip_runtime.h>
#include <math.h>

#define C_CH   64
#define HID    64
#define KE     9
#define HH     128
#define WW     128
#define BATCH  4
#define OCH    (C_CH * KE)   // 576
#define NSTEPS 5

#define TH 8
#define TW 16
#define HALO_F (C_CH * (TH + 2) * (TW + 2))  // 64*10*18 = 11520 floats
#define KBUF_F (144 * (TH * TW))             // 144*128  = 18432 floats

typedef _Float16 half_t;
typedef __attribute__((ext_vector_type(16))) _Float16 v16h;
typedef __attribute__((ext_vector_type(8)))  float    v8f;

// A-matrix fragment (16x32 f16, row-major source, row stride 64 halves, K base = 32*kk)
// lane<16: K = {0..15} (+32*kk); lane>=16: K = {16..31} (+32*kk)
__device__ __forceinline__ v16h load_a_frag(const half_t* __restrict__ arow, int hs, int kk) {
  v16h a;
  const half_t* p = arow + 32 * kk + 16 * hs;
  #pragma unroll
  for (int i = 0; i < 8; ++i) { a[i] = p[i]; a[8 + i] = p[8 + i]; }
  return a;
}

// ---------------- Kernel 0: convert weights to f16 once ----------------
__global__ void __launch_bounds__(256)
k_prep(const float* __restrict__ W1, const float* __restrict__ W2,
       half_t* __restrict__ W1h, half_t* __restrict__ W2h) {
  int i = blockIdx.x * 256 + threadIdx.x;
  if (i < HID * C_CH) W1h[i] = (half_t)W1[i];
  if (i < OCH * HID)  W2h[i] = (half_t)W2[i];
}

// ---------------- Kernel 1: t = GELU(W1 @ h + b1), pixel-major f16 ----------------
// grid = (HH, BATCH), block = 256 (8 waves). Each WG: one image row (128 pixels).
// WMMA: M = hid (4 tiles), N = 16 pixels per wave, K = C (2 k-steps).
__global__ void __launch_bounds__(256)
k_pred1(const float* __restrict__ hcur, const half_t* __restrict__ W1h,
        const float* __restrict__ b1, half_t* __restrict__ t) {
  __shared__ float hl[C_CH * WW];  // 32 KB: h[c][x] for this row
  const int b = blockIdx.y, y = blockIdx.x;
  const float* hbase = hcur + (size_t)b * C_CH * HH * WW + (size_t)y * WW;
  for (int idx = threadIdx.x; idx < C_CH * WW; idx += 256) {
    int c = idx >> 7, x = idx & 127;
    hl[idx] = hbase[(size_t)c * HH * WW + x];   // coalesced per channel row
  }
  __syncthreads();

  const int lane = threadIdx.x & 31, wav = threadIdx.x >> 5;
  const int hs = lane >> 4, n = lane & 15;
  const int x = wav * 16 + n;

  // B fragments (32x16): lanes 0-15 hold K=0..15, lanes 16-31 hold K=16..31
  v16h bf0, bf1;
  #pragma unroll
  for (int i = 0; i < 16; ++i) {
    int kl = 16 * hs + i;
    bf0[i] = (half_t)hl[kl * WW + x];          // K = 0..31
    bf1[i] = (half_t)hl[(32 + kl) * WW + x];   // K = 32..63
  }

  half_t* trow = t + ((size_t)(b * HH + y) * WW + x) * HID;
  #pragma unroll
  for (int mt = 0; mt < 4; ++mt) {
    const half_t* arow = W1h + (size_t)(mt * 16 + (lane & 15)) * HID;
    v16h a0 = load_a_frag(arow, hs, 0);
    v16h a1 = load_a_frag(arow, hs, 1);
    v8f d = {};
    d = __builtin_amdgcn_wmma_f32_16x16x32_f16(false, a0, false, bf0, (short)0, d, false, false);
    d = __builtin_amdgcn_wmma_f32_16x16x32_f16(false, a1, false, bf1, (short)0, d, false, false);
    #pragma unroll
    for (int v = 0; v < 8; ++v) {
      int hid = mt * 16 + v + 8 * hs;
      float val = d[v] + b1[hid];
      val = 0.5f * val * (1.0f + erff(val * 0.70710678118654752f));  // exact GELU
      trow[hid] = (half_t)val;
    }
  }
}

// ---------------- Kernel 2: fused k = W2@t + b2, L2-normalize(KE), 3x3 dyn conv, residual ----
// grid = (WW/TW, HH/TH, BATCH), block = 256 (8 waves). Tile = 8x16 pixels, all 64 channels.
// WMMA: M = output rows (c*9+e) within a 16-channel group (9 tiles), N = 16 pixels (1 N-tile/wave).
__global__ void __launch_bounds__(256)
k_dynconv(const float* __restrict__ hcur, const half_t* __restrict__ W2h,
          const float* __restrict__ b2, const half_t* __restrict__ t,
          const float* __restrict__ alphap, float* __restrict__ hnext) {
  extern __shared__ float smem[];
  float* halo = smem;            // [64][10][18]
  float* kbuf = smem + HALO_F;   // [144][128]

  const int b  = blockIdx.z;
  const int y0 = blockIdx.y * TH;
  const int x0 = blockIdx.x * TW;
  const float alpha = alphap[0];

  // Load h tile + 1-pixel halo (zero padded at borders)
  for (int idx = threadIdx.x; idx < HALO_F; idx += 256) {
    int c = idx / 180; int rem = idx - c * 180;
    int r = rem / 18;  int col = rem - r * 18;
    int gy = y0 + r - 1, gx = x0 + col - 1;
    float v = 0.0f;
    if (gy >= 0 && gy < HH && gx >= 0 && gx < WW)
      v = hcur[(((size_t)b * C_CH + c) * HH + gy) * WW + gx];
    halo[idx] = v;
  }

  const int lane = threadIdx.x & 31, wav = threadIdx.x >> 5;
  const int hs = lane >> 4, n = lane & 15;
  const int yy = y0 + wav, xx = x0 + n;

  // B fragments from t (contiguous 16-half runs per lane; reused for all 36 M-tiles)
  const half_t* trow = t + ((size_t)(b * HH + yy) * WW + xx) * HID;
  v16h bf0, bf1;
  #pragma unroll
  for (int i = 0; i < 16; ++i) {
    bf0[i] = trow[16 * hs + i];
    bf1[i] = trow[32 + 16 * hs + i];
  }
  __syncthreads();

  for (int g = 0; g < 4; ++g) {          // channel groups of 16 -> 144 output rows
    #pragma unroll
    for (int mt = 0; mt < 9; ++mt) {
      const half_t* arow = W2h + (size_t)(g * 144 + mt * 16 + (lane & 15)) * HID;
      v16h a0 = load_a_frag(arow, hs, 0);
      v16h a1 = load_a_frag(arow, hs, 1);
      v8f d = {};
      d = __builtin_amdgcn_wmma_f32_16x16x32_f16(false, a0, false, bf0, (short)0, d, false, false);
      d = __builtin_amdgcn_wmma_f32_16x16x32_f16(false, a1, false, bf1, (short)0, d, false, false);
      #pragma unroll
      for (int v = 0; v < 8; ++v) {
        int ol = mt * 16 + v + 8 * hs;
        kbuf[ol * (TH * TW) + wav * 16 + n] = d[v] + b2[g * 144 + ol];
      }
    }
    __syncthreads();

    // Normalize over KE and apply dynamic depthwise conv + residual
    for (int idx = threadIdx.x; idx < 16 * TH * TW; idx += 256) {
      int cl = idx >> 7, pix = idx & 127;
      int r = pix >> 4, xl = pix & 15;
      float kv[9]; float ss = 0.0f;
      #pragma unroll
      for (int e = 0; e < 9; ++e) {
        float v = kbuf[(cl * KE + e) * (TH * TW) + pix];
        kv[e] = v; ss += v * v;
      }
      float inv = 1.0f / (sqrtf(ss) + 1e-6f);
      const float* hc = halo + (g * 16 + cl) * 180;
      float acc = 0.0f;
      #pragma unroll
      for (int di = 0; di < 3; ++di)
        #pragma unroll
        for (int dj = 0; dj < 3; ++dj)
          acc += kv[di * 3 + dj] * hc[(r + di) * 18 + (xl + dj)];
      float center = hc[(r + 1) * 18 + (xl + 1)];
      hnext[(((size_t)b * C_CH + g * 16 + cl) * HH + (y0 + r)) * WW + (x0 + xl)] =
          center + alpha * inv * acc;
    }
    __syncthreads();  // kbuf reused by next group
  }
}

extern "C" void kernel_launch(void* const* d_in, const int* in_sizes, int n_in,
                              void* d_out, int out_size, void* d_ws, size_t ws_size,
                              hipStream_t stream) {
  const float* x     = (const float*)d_in[0];
  const float* W1    = (const float*)d_in[1];
  const float* b1    = (const float*)d_in[2];
  const float* W2    = (const float*)d_in[3];
  const float* b2    = (const float*)d_in[4];
  const float* alpha = (const float*)d_in[5];

  char* ws = (char*)d_ws;
  half_t* W1h = (half_t*)ws;                        // 8 KB
  half_t* W2h = (half_t*)(ws + 8192);               // 72 KB
  half_t* t   = (half_t*)(ws + 8192 + 73728);       // 8 MB (B*H*W*HID f16)
  size_t off  = 8192 + 73728 + (size_t)BATCH * HH * WW * HID * sizeof(half_t);
  float* hA   = (float*)(ws + off);                                               // 16 MB
  float* hB   = (float*)(ws + off + (size_t)BATCH * C_CH * HH * WW * sizeof(float)); // 16 MB

  k_prep<<<(OCH * HID + 255) / 256, 256, 0, stream>>>(W1, W2, W1h, W2h);

  const size_t smem2 = (HALO_F + KBUF_F) * sizeof(float);  // 119808 B (< 320 KB WGP LDS)
  const float* cur = x;
  for (int s = 0; s < NSTEPS; ++s) {
    float* nxt = (s == NSTEPS - 1) ? (float*)d_out : ((s & 1) ? hB : hA);
    k_pred1<<<dim3(HH, BATCH), 256, 0, stream>>>(cur, W1h, b1, t);
    k_dynconv<<<dim3(WW / TW, HH / TH, BATCH), 256, smem2, stream>>>(cur, W2h, b2, t, alpha, nxt);
    cur = nxt;
  }
}
